// RNN_88862873354365
// MI455X (gfx1250) — compile-verified, matrix operand order checked by reading, not run on hardware
//
#include <hip/hip_runtime.h>
#include <hip/hip_bf16.h>

// Problem sizes (fixed by the reference)
#define BB 32
#define SS 2048
#define VV 512
#define HH 512
#define OO 512

typedef __attribute__((ext_vector_type(16))) _Float16 v16h;
typedef __attribute__((ext_vector_type(8)))  _Float16 v8h;
typedef __attribute__((ext_vector_type(8)))  float    v8f;

// CDNA5 16-bit A/B fragment layout (ISA 7.12.2), per lane (half = lane>>4):
//   elems 0..7  -> K = k0 + half*8 + e        (8 contiguous values)
//   elems 8..15 -> K = k0 + 16 + half*8 + e   (8 contiguous values)
__device__ __forceinline__ v16h combine8(v8h lo, v8h hi) {
  return __builtin_shufflevector(lo, hi, 0, 1, 2, 3, 4, 5, 6, 7,
                                 8, 9, 10, 11, 12, 13, 14, 15);
}

// f32 source -> f16 fragment (used once for the Whh register panel)
__device__ __forceinline__ v16h load_frag_f32(const float* __restrict__ p) {
  v16h f;
#pragma unroll
  for (int e = 0; e < 8; ++e) f[e] = (_Float16)p[e];
#pragma unroll
  for (int e = 0; e < 8; ++e) f[e + 8] = (_Float16)p[e + 16];
  return f;
}

// f16 source (16B aligned) -> fragment: two b128 loads, no VALU
__device__ __forceinline__ v16h load_frag_f16(const _Float16* __restrict__ p) {
  return combine8(*(const v8h*)p, *(const v8h*)(p + 16));
}

// ---------------------------------------------------------------------------
// One-time f32 -> f16 weight conversion (Wxh, Who) into d_ws
// ---------------------------------------------------------------------------
__global__ void cvt_w_f16(const float* __restrict__ src,
                          _Float16* __restrict__ dst, int n) {
  for (int i = blockIdx.x * blockDim.x + threadIdx.x; i < n;
       i += gridDim.x * blockDim.x)
    dst[i] = (_Float16)src[i];
}

// ---------------------------------------------------------------------------
// Phases 1 & 3: Out[M,512] = A[M,512] @ W16[512,512]^T + bias.
// Block = 256 threads (8 waves) covers a 32-row A tile, staged once in LDS as
// f16 (padded rows: +16 f16 = 8-dword bank skew). Each wave owns 16M x 128N.
// A read exactly once from HBM; W16 fragments are 2x b128 f16 loads (L2-hot).
// ---------------------------------------------------------------------------
#define APAD (512 + 16)
__global__ __launch_bounds__(256) void gemm_bias_f16wmma(
    const float* __restrict__ A,
    const _Float16* __restrict__ W16,
    const float* __restrict__ bias,
    float* __restrict__ Out)
{
  __shared__ _Float16 atile[32][APAD];
  const int tid  = threadIdx.x;
  const int lane = tid & 31;
  const int wv   = tid >> 5;          // 0..7
  const int m0   = blockIdx.x * 32;

  // Cooperative stage: 32x512 f32 -> f16 in LDS (8 elems per thread per iter)
  for (int i = tid; i < (32 * 512) / 8; i += 256) {
    const int idx = i * 8;
    const int row = idx >> 9;
    const int col = idx & 511;
    const float* gp = A + (size_t)(m0 + row) * 512 + col;
    v8h h;
#pragma unroll
    for (int e = 0; e < 8; ++e) h[e] = (_Float16)gp[e];
    *(v8h*)&atile[row][col] = h;
  }
  __syncthreads();

  const int mt   = wv >> 2;           // 0..1 : which 16-row half
  const int ngrp = wv & 3;            // 0..3 : which 128-col group
  const int n0   = ngrp * 128;
  const int mr   = lane & 15;         // M for A / N for B,C,D
  const int half = lane >> 4;

  const v8f vzero = {};
  v8f acc[8];
#pragma unroll
  for (int t = 0; t < 8; ++t) acc[t] = vzero;

  const _Float16* arow = &atile[mt * 16 + mr][0];

  for (int k0 = 0; k0 < 512; k0 += 32) {
    v16h af = load_frag_f16(arow + k0 + half * 8);
#pragma unroll
    for (int t = 0; t < 8; ++t) {
      const _Float16* bp =
          W16 + (size_t)(n0 + t * 16 + mr) * 512 + k0 + half * 8;
      v16h bf = load_frag_f16(bp);
      acc[t] = __builtin_amdgcn_wmma_f32_16x16x32_f16(
          false, af, false, bf, (short)0, acc[t], false, false);
    }
  }

  // C/D: lane&15 = N; VGPR r -> M = r + (lane>>4)*8
#pragma unroll
  for (int t = 0; t < 8; ++t) {
    const int col = n0 + t * 16 + mr;
    const float bv = bias[col];
#pragma unroll
    for (int r = 0; r < 8; ++r) {
      const int row = m0 + mt * 16 + r + half * 8;
      Out[(size_t)row * 512 + col] = acc[t][r] + bv;
    }
  }
}

// ---------------------------------------------------------------------------
// Phase 2: H_s = xp_s + H_{s-1} @ Whh^T + bhh, 2048 sequential steps.
// Batch rows independent -> 2 workgroups x 16 rows; 32 waves each own one
// 16-col N-tile with their 512x16 Whh^T slice in registers (16 x v16h).
// H state double-buffered in LDS as f16 (fragment loads = 2x ds_load_b128,
// no conversions on the serial path). XP updated in place -> Hs (f32).
// ---------------------------------------------------------------------------
#define HPAD (HH + 16)
__global__ __launch_bounds__(1024) void rnn_recurrence(
    float* __restrict__ XP,        // [B,S,H] xproj -> overwritten with Hs
    const float* __restrict__ Whh, // [H,H] f32
    const float* __restrict__ bhh, // [H]
    float* __restrict__ Hlast)     // [B,H]
{
  __shared__ _Float16 hbuf[2][16][HPAD];
  const int lane = threadIdx.x & 31;
  const int wave = threadIdx.x >> 5;   // 0..31 -> N-tile
  const int n0   = wave * 16;
  const int b0   = blockIdx.x * 16;    // batch-row block
  const int nn   = lane & 15;          // N (B/C/D) and M (A) lane index
  const int half = lane >> 4;

  // H0 = ones (pad lanes harmless)
  for (int i = threadIdx.x; i < 16 * HPAD; i += blockDim.x)
    (&hbuf[0][0][0])[i] = (_Float16)1.0f;

  // Register-resident B panel: B[k, n] = Whh[n, k]
  v16h bpanel[16];
#pragma unroll
  for (int kc = 0; kc < 16; ++kc)
    bpanel[kc] =
        load_frag_f32(Whh + (size_t)(n0 + nn) * HH + kc * 32 + half * 8);

  const float bv = bhh[n0 + nn];
  __syncthreads();

  for (int s = 0; s < SS; ++s) {
    const int cur = s & 1, nxt = cur ^ 1;
    v8f acc = {};
#pragma unroll
    for (int kc = 0; kc < 16; ++kc) {
      v16h af = load_frag_f16(&hbuf[cur][nn][kc * 32 + half * 8]);
      acc = __builtin_amdgcn_wmma_f32_16x16x32_f16(
          false, af, false, bpanel[kc], (short)0, acc, false, false);
    }
#pragma unroll
    for (int r = 0; r < 8; ++r) {
      const int row = r + half * 8;                       // local batch row
      const size_t g = (((size_t)(b0 + row) * SS) + s) * HH + n0 + nn;
      const float v = XP[g] + acc[r] + bv;                // xp + H@Whh^T + bhh
      XP[g] = v;                                          // becomes Hs (f32)
      hbuf[nxt][row][n0 + nn] = (_Float16)v;              // state for s+1
      __builtin_prefetch(XP + g + HH, 0, 0);              // next step's xp
      if (s == SS - 1)
        Hlast[(size_t)(b0 + row) * HH + n0 + nn] = v;     // full-f32 Hlast
    }
    __syncthreads();
  }
}

extern "C" void kernel_launch(void* const* d_in, const int* in_sizes, int n_in,
                              void* d_out, int out_size, void* d_ws, size_t ws_size,
                              hipStream_t stream) {
  (void)in_sizes; (void)n_in; (void)out_size; (void)ws_size;
  const float* X   = (const float*)d_in[0];
  const float* Wxh = (const float*)d_in[1];
  const float* bxh = (const float*)d_in[2];
  const float* Whh = (const float*)d_in[3];
  const float* bhh = (const float*)d_in[4];
  const float* Who = (const float*)d_in[5];
  const float* bho = (const float*)d_in[6];

  float* Y     = (float*)d_out;                    // [B,S,O]
  float* Hlast = Y + (size_t)BB * SS * OO;         // [B,H]

  char* ws = (char*)d_ws;
  float*     XP   = (float*)ws;                              // [B,S,H] f32
  _Float16*  W16x = (_Float16*)(ws + (size_t)BB * SS * HH * sizeof(float));
  _Float16*  W16o = W16x + (size_t)HH * VV;

  const int M = BB * SS;                           // 65536 rows
  const int blocks = M / 32;                       // 32-row tile per block

  // One-time weight conversion to f16
  cvt_w_f16<<<256, 256, 0, stream>>>(Wxh, W16x, HH * VV);
  cvt_w_f16<<<256, 256, 0, stream>>>(Who, W16o, OO * HH);
  // Phase 1: xproj = X @ Wxh^T + bxh   -> XP
  gemm_bias_f16wmma<<<blocks, 256, 0, stream>>>(X, W16x, bxh, XP);
  // Phase 2: sequential scan, in place XP -> Hs, plus Hlast
  rnn_recurrence<<<2, 1024, 0, stream>>>(XP, Whh, bhh, Hlast);
  // Phase 3: Y = Hs @ Who^T + bho
  gemm_bias_f16wmma<<<blocks, 256, 0, stream>>>(XP, W16o, bho, Y);
}
